// ConditionedPNA_63763084477013
// MI455X (gfx1250) — matrix operand order repaired and checked by reading, not compiled, not run on hardware
//
#include <hip/hip_runtime.h>
#include <hip/hip_bf16.h>
#include <math.h>

// ---------------- problem constants ----------------
#define BQ      2
#define NNODES  20000
#define NTOT    (BQ * NNODES)        // 40000
#define EDIR    320000
#define EPG     (2 * EDIR)           // 640000 edges per graph (both directions)
#define ETOT    (BQ * EPG)           // 1,280,000
#define MTEXT   5000
#define DD      64
#define NREL    20
#define NLAYERS 2
#define KS      4000                 // top-k nodes per batch
#define ES      128000               // top-k edges per batch
#define NEGF    (-3.4028234663852886e38f)

typedef __attribute__((ext_vector_type(16))) _Float16 v16h;
typedef __attribute__((ext_vector_type(8)))  float    v8f;

// orderable-uint mapping for float compare / atomic min-max
__device__ __forceinline__ unsigned ordf(float f) {
    unsigned u = __float_as_uint(f);
    return (u & 0x80000000u) ? ~u : (u | 0x80000000u);
}
__device__ __forceinline__ float unordf(unsigned u) {
    return (u & 0x80000000u) ? __uint_as_float(u & 0x7fffffffu)
                             : __uint_as_float(~u);
}

// decode flattened edge id -> (batch, global src, global dst, per-graph edge idx)
__device__ __forceinline__ void edge_decode(const int* __restrict__ ei, int e,
                                            int& b, int& gs, int& gd, int& eidx) {
    b = e / EPG;
    int ee = e - b * EPG;
    int dir = ee >= EDIR;
    eidx = dir ? (ee - EDIR) : ee;
    int s0 = dir ? ei[EDIR + eidx] : ei[eidx];
    int d0 = dir ? ei[eidx] : ei[EDIR + eidx];
    gs = s0 + b * NNODES;
    gd = d0 + b * NNODES;
}

// ---------------- pack f32 weight (K x N, row major) into f16 WMMA-B lane layout ----
// out[(((nt*chunks)+kc)*32 + lane)*16 + j]  holds W[k][n] with:
//   hi = lane>>4, n = nt*16 + (lane&15), k = kc*32 + hi*8 + (j&7) + (j>>3)*16
__global__ void kpackB(const float* __restrict__ W, int K, int N,
                       _Float16* __restrict__ out) {
    int idx = blockIdx.x * blockDim.x + threadIdx.x;
    if (idx >= K * N) return;
    int chunks = K >> 5;
    int j = idx & 15;
    int lane = (idx >> 4) & 31;
    int rest = idx >> 9;
    int kc = rest % chunks;
    int nt = rest / chunks;
    int hi = lane >> 4;
    int nn = nt * 16 + (lane & 15);
    int k = kc * 32 + hi * 8 + (j & 7) + (j >> 3) * 16;
    out[idx] = (_Float16)W[k * N + nn];
}

// ---------------- prep: h2/t2/r2, query rows ----------------
__global__ void kprep(const int* __restrict__ h_index, const int* __restrict__ t_index,
                      const int* __restrict__ r_index, const float* __restrict__ rel_embedding,
                      int* __restrict__ h2_0, int* __restrict__ t2, float* __restrict__ rel_q) {
    __shared__ int neq;
    int b = blockIdx.x;
    if (threadIdx.x == 0) neq = 0;
    __syncthreads();
    if (threadIdx.x < 16) {
        if (h_index[b * 16 + threadIdx.x] != h_index[b * 16]) atomicOr(&neq, 1);
    }
    __syncthreads();
    bool itn = (neq == 0);                                 // is_t_neg
    if (threadIdx.x < 16) {
        int j = threadIdx.x;
        int tv = (itn ? t_index[b * 16 + j] : h_index[b * 16 + j]) + b * NNODES;
        t2[b * 16 + j] = tv;
    }
    if (threadIdx.x == 0) {
        h2_0[b] = (itn ? h_index[b * 16] : t_index[b * 16]) + b * NNODES;
    }
    int r2 = itn ? r_index[b * 16] : (r_index[b * 16] + NREL);
    for (int d = threadIdx.x; d < DD; d += 32)
        rel_q[b * DD + d] = rel_embedding[r2 * DD + d];
}

// ---------------- scatter text embeddings / hidden_states ----------------
__global__ void kscatter_text(const int* __restrict__ all_index,
                              const float* __restrict__ text, float* __restrict__ hidden) {
    int m = blockIdx.x;
    int node = all_index[m];
    hidden[node * DD + threadIdx.x] = text[m * DD + threadIdx.x];
}
__global__ void kscatter_h(const int* __restrict__ h2_0, const float* __restrict__ hs,
                           float* __restrict__ hidden) {
    int b = blockIdx.x;
    hidden[h2_0[b] * DD + threadIdx.x] = hs[b * DD + threadIdx.x];
}

// ---------------- initial score constants per batch ----------------
__global__ __launch_bounds__(128) void kinit_sc(
        const float* __restrict__ hs, const float* __restrict__ rel_q,
        const float* __restrict__ linW, const float* __restrict__ linb,
        const float* __restrict__ W1, const float* __restrict__ b1,
        const float* __restrict__ W2, const float* __restrict__ b2,
        float* __restrict__ s0, float* __restrict__ sh) {
    int b = blockIdx.x;
    __shared__ float xs[DD];
    __shared__ float redl[128];
    for (int pass = 0; pass < 2; ++pass) {
        int d = threadIdx.x;
        if (d < DD) {
            float hv = pass ? hs[b * DD + d] : 0.0f;
            float hsum = linb[d];
            for (int k = 0; k < DD; ++k) {
                float hk = pass ? hs[b * DD + k] : 0.0f;
                hsum += hk * linW[k * DD + d];
            }
            for (int k = 0; k < DD; ++k)
                hsum += rel_q[b * DD + k] * linW[(DD + k) * DD + d];
            xs[d] = hv * hsum;
        }
        __syncthreads();
        int j = threadIdx.x;
        float s = b1[j];
        for (int dd = 0; dd < DD; ++dd) s += xs[dd] * W1[dd * 128 + j];
        redl[j] = fmaxf(s, 0.0f) * W2[j];
        __syncthreads();
        for (int st = 64; st > 0; st >>= 1) {
            if (threadIdx.x < st) redl[threadIdx.x] += redl[threadIdx.x + st];
            __syncthreads();
        }
        if (threadIdx.x == 0) {
            float out = redl[0] + b2[0];
            if (pass) sh[b] = out; else s0[b] = out;
        }
        __syncthreads();
    }
}

__global__ void kfill_score(const int* __restrict__ h2_0, const float* __restrict__ s0,
                            const float* __restrict__ sh, float* __restrict__ score) {
    int i = blockIdx.x * blockDim.x + threadIdx.x;
    if (i >= NTOT) return;
    int b = i / NNODES;
    score[i] = (i == h2_0[b]) ? sh[b] : s0[b];
}

// ---------------- degree + delta ----------------
__global__ void kdeg(const int* __restrict__ ei, float* __restrict__ deg) {
    int e = blockIdx.x * blockDim.x + threadIdx.x;
    if (e >= ETOT) return;
    int b, gs, gd, eidx;
    edge_decode(ei, e, b, gs, gd, eidx);
    atomicAdd(&deg[gs], 1.0f);
}
__global__ __launch_bounds__(256) void kdelta(const float* __restrict__ deg,
                                              float* __restrict__ delta) {
    __shared__ float rs[256];
    float s = 0.0f;
    for (int i = threadIdx.x; i < NTOT; i += 256) s += logf(deg[i] + 1.0f);
    rs[threadIdx.x] = s;
    __syncthreads();
    for (int st = 128; st > 0; st >>= 1) {
        if (threadIdx.x < st) rs[threadIdx.x] += rs[threadIdx.x + st];
        __syncthreads();
    }
    if (threadIdx.x == 0) *delta = rs[0] / (float)NTOT;
}

// ---------------- radix-select top-k threshold: nodes ----------------
__global__ __launch_bounds__(256) void knode_topk(const float* __restrict__ score,
                                                  unsigned* __restrict__ nthr) {
    int b = blockIdx.x;
    __shared__ unsigned hist[256];
    __shared__ unsigned sprefix;
    __shared__ int swant;
    if (threadIdx.x == 0) { sprefix = 0u; swant = KS; }
    __syncthreads();
    for (int byte = 3; byte >= 0; --byte) {
        hist[threadIdx.x] = 0u;
        __syncthreads();
        unsigned prefix = sprefix;
        for (int i = threadIdx.x; i < NNODES; i += 256) {
            unsigned u = ordf(score[b * NNODES + i]);
            bool match = (byte == 3) ||
                         ((u >> ((byte + 1) * 8)) == (prefix >> ((byte + 1) * 8)));
            if (match) atomicAdd(&hist[(u >> (byte * 8)) & 255u], 1u);
        }
        __syncthreads();
        if (threadIdx.x == 0) {
            int want = swant, cum = 0;
            for (int bin = 255; bin >= 0; --bin) {
                int c = (int)hist[bin];
                if (cum + c >= want) {
                    sprefix = prefix | ((unsigned)bin << (byte * 8));
                    swant = want - cum;
                    break;
                }
                cum += c;
            }
        }
        __syncthreads();
    }
    if (threadIdx.x == 0) nthr[b] = sprefix;
}

__global__ void knode_sel(const float* __restrict__ score, const unsigned* __restrict__ nthr,
                          unsigned char* __restrict__ node_sel) {
    int i = blockIdx.x * blockDim.x + threadIdx.x;
    if (i >= NTOT) return;
    int b = i / NNODES;
    node_sel[i] = (ordf(score[i]) >= nthr[b]) ? 1 : 0;
}

// ---------------- radix-select top-k threshold: edges ----------------
__device__ __forceinline__ float edge_val(const int* __restrict__ ei, int b, int ee,
                                          const unsigned char* __restrict__ node_sel,
                                          const float* __restrict__ score) {
    int dir = ee >= EDIR;
    int eidx = dir ? (ee - EDIR) : ee;
    int s0 = dir ? ei[EDIR + eidx] : ei[eidx];
    int d0 = dir ? ei[eidx] : ei[EDIR + eidx];
    int gs = s0 + b * NNODES, gd = d0 + b * NNODES;
    return node_sel[gs] ? score[gd] : NEGF;
}

__global__ __launch_bounds__(512) void kedge_topk(const int* __restrict__ ei,
                                                  const unsigned char* __restrict__ node_sel,
                                                  const float* __restrict__ score,
                                                  unsigned* __restrict__ eth) {
    int b = blockIdx.x;
    __shared__ unsigned hist[256];
    __shared__ unsigned sprefix;
    __shared__ int swant;
    if (threadIdx.x == 0) { sprefix = 0u; swant = ES; }
    __syncthreads();
    for (int byte = 3; byte >= 0; --byte) {
        if (threadIdx.x < 256) hist[threadIdx.x] = 0u;
        __syncthreads();
        unsigned prefix = sprefix;
        for (int i = threadIdx.x; i < EPG; i += 512) {
            unsigned u = ordf(edge_val(ei, b, i, node_sel, score));
            bool match = (byte == 3) ||
                         ((u >> ((byte + 1) * 8)) == (prefix >> ((byte + 1) * 8)));
            if (match) atomicAdd(&hist[(u >> (byte * 8)) & 255u], 1u);
        }
        __syncthreads();
        if (threadIdx.x == 0) {
            int want = swant, cum = 0;
            for (int bin = 255; bin >= 0; --bin) {
                int c = (int)hist[bin];
                if (cum + c >= want) {
                    sprefix = prefix | ((unsigned)bin << (byte * 8));
                    swant = want - cum;
                    break;
                }
                cum += c;
            }
        }
        __syncthreads();
    }
    if (threadIdx.x == 0) eth[b] = sprefix;
}

__global__ void kedge_mask(const int* __restrict__ ei, const unsigned char* __restrict__ node_sel,
                           const float* __restrict__ score, const unsigned* __restrict__ eth,
                           unsigned char* __restrict__ edge_sel) {
    int e = blockIdx.x * blockDim.x + threadIdx.x;
    if (e >= ETOT) return;
    int b = e / EPG;
    int ee = e - b * EPG;
    float v = edge_val(ei, b, ee, node_sel, score);
    edge_sel[e] = (v > NEGF && ordf(v) >= eth[b]) ? 1 : 0;
}

// ---------------- layer input = sigmoid(score) * hidden ----------------
__global__ void klayer_input(const float* __restrict__ score, const float* __restrict__ hidden,
                             float* __restrict__ li) {
    int i = blockIdx.x * blockDim.x + threadIdx.x;
    if (i >= NTOT * DD) return;
    float s = score[i >> 6];
    li[i] = hidden[i] * (1.0f / (1.0f + expf(-s)));
}

// ---------------- masked edge aggregation (64 threads per edge) ----------------
__global__ __launch_bounds__(256) void kedge_agg(
        const int* __restrict__ ei, const int* __restrict__ edge_attr,
        const unsigned char* __restrict__ edge_sel, const float* __restrict__ li,
        const float* __restrict__ crel,    // conv_rel_emb for this layer
        float* __restrict__ mean_sum, float* __restrict__ sq_sum,
        unsigned* __restrict__ mx_u, unsigned* __restrict__ mn_u,
        float* __restrict__ cnt_dst, float* __restrict__ cnt_src) {
    int e = blockIdx.x * 4 + (threadIdx.x >> 6);
    int d = threadIdx.x & 63;
    if (e >= ETOT || !edge_sel[e]) return;
    int b, gs, gd, eidx;
    edge_decode(ei, e, b, gs, gd, eidx);
    int rel = edge_attr[eidx];
    float msg = li[gs * DD + d] + crel[rel * DD + d];
    atomicAdd(&mean_sum[gd * DD + d], msg);
    atomicAdd(&sq_sum[gd * DD + d], msg * msg);
    atomicMax(&mx_u[gd * DD + d], ordf(msg));
    atomicMin(&mn_u[gd * DD + d], ordf(msg));
    if (d == 0) {
        atomicAdd(&cnt_dst[gd], 1.0f);
        atomicAdd(&cnt_src[gs], 1.0f);
    }
}

__global__ void kupd(const float* __restrict__ cnt_src, const float* __restrict__ cnt_dst,
                     const float* __restrict__ deg, unsigned char* __restrict__ upd) {
    int i = blockIdx.x * blockDim.x + threadIdx.x;
    if (i >= NTOT) return;
    upd[i] = ((cnt_src[i] + cnt_dst[i]) > 0.0f && deg[i] > 0.0f) ? 1 : 0;
}

// ---------------- PNA conv: feat(13D=832) @ W(832x64) via WMMA ----------------
__global__ __launch_bounds__(128) void kconv(
        const float* __restrict__ li, const float* __restrict__ mean_sum,
        const float* __restrict__ sq_sum, const unsigned* __restrict__ mx_u,
        const unsigned* __restrict__ mn_u, const float* __restrict__ cnt_dst,
        const float* __restrict__ deg, const float* __restrict__ delta_p,
        const unsigned char* __restrict__ upd,
        const _Float16* __restrict__ Wp,   // packed B layout, chunks=26, ntiles=4
        const float* __restrict__ convb, float* __restrict__ hidden) {
    __shared__ _Float16 As[16][13 * DD];   // 16 x 832 f16 = 26 KB
    __shared__ unsigned char upds[16];
    __shared__ int anyupd;
    int base = blockIdx.x * 16;
    if (threadIdx.x == 0) anyupd = 0;
    __syncthreads();
    if (threadIdx.x < 16) {
        unsigned char u = upd[base + threadIdx.x];
        upds[threadIdx.x] = u;
        if (u) atomicOr(&anyupd, 1);
    }
    __syncthreads();
    if (!anyupd) return;                   // tile fully gated off: exact skip
    float delta = *delta_p;
    for (int idx = threadIdx.x; idx < 16 * DD; idx += 128) {
        int r = idx >> 6, d = idx & 63;
        int node = base + r;
        float cd = cnt_dst[node];
        float denom = fmaxf(cd, 1.0f);
        float mean = mean_sum[node * DD + d] / denom;
        float sq   = sq_sum[node * DD + d] / denom;
        float stdv = sqrtf(fmaxf(sq - mean * mean, 0.0f) + 1e-10f);
        bool has = cd > 0.0f;
        float mx = has ? unordf(mx_u[node * DD + d]) : 0.0f;
        float mn = has ? unordf(mn_u[node * DD + d]) : 0.0f;
        float dl = logf(deg[node] + 1.0f);
        float samp = dl / delta;
        float satt = (dl > 0.0f) ? (delta / fmaxf(dl, 1e-6f)) : 0.0f;
        As[r][d]       = (_Float16)li[node * DD + d];
        As[r][64 + d]  = (_Float16)mean;
        As[r][128 + d] = (_Float16)mx;
        As[r][192 + d] = (_Float16)mn;
        As[r][256 + d] = (_Float16)stdv;
        As[r][320 + d] = (_Float16)(mean * samp);
        As[r][384 + d] = (_Float16)(mx * samp);
        As[r][448 + d] = (_Float16)(mn * samp);
        As[r][512 + d] = (_Float16)(stdv * samp);
        As[r][576 + d] = (_Float16)(mean * satt);
        As[r][640 + d] = (_Float16)(mx * satt);
        As[r][704 + d] = (_Float16)(mn * satt);
        As[r][768 + d] = (_Float16)(stdv * satt);
    }
    __syncthreads();
    int lane = threadIdx.x & 31, wv = threadIdx.x >> 5;
    int hi = lane >> 4, mrow = lane & 15;
    int nn = wv * 16 + (lane & 15);
    v8f c = {};
    for (int kc = 0; kc < 26; ++kc) {
        int K0 = kc * 32;
        v16h a;
#pragma unroll
        for (int v = 0; v < 8; ++v) {
            int ka = K0 + ((v < 4) ? 2 * v : 16 + 2 * (v - 4)) + hi * 8;
            a[2 * v]     = As[mrow][ka];
            a[2 * v + 1] = As[mrow][ka + 1];
        }
        v16h w = *(const v16h*)(Wp + (((size_t)wv * 26 + kc) * 32 + lane) * 16);
        c = __builtin_amdgcn_wmma_f32_16x16x32_f16(false, a, false, w,
                                                   (short)0, c, false, false);
    }
    float bias = convb[nn];
#pragma unroll
    for (int r = 0; r < 8; ++r) {
        int m = r + hi * 8;
        if (upds[m]) {
            float h = fmaxf(c[r] + bias, 0.0f);
            hidden[(base + m) * DD + nn] += h;
        }
    }
}

// ---------------- score MLP for all nodes via WMMA chain ----------------
__global__ __launch_bounds__(128) void kscore(
        const float* __restrict__ hidden, const float* __restrict__ rel_q,
        const _Float16* __restrict__ linWp,  // packed, chunks=4, ntiles=4
        const float* __restrict__ linb,
        const _Float16* __restrict__ W1p,    // packed, chunks=2, ntiles=8
        const float* __restrict__ b1,
        const float* __restrict__ W2, const float* __restrict__ b2,
        const unsigned char* __restrict__ upd, float* __restrict__ score) {
    __shared__ _Float16 A1[16][128];
    __shared__ float    Heur[16][64];
    __shared__ _Float16 A2[16][64];
    __shared__ float    H1[16][128];
    __shared__ float    red[16][8];
    __shared__ unsigned char upds[16];
    __shared__ int anyupd;
    int base = blockIdx.x * 16;
    if (threadIdx.x == 0) anyupd = 0;
    __syncthreads();
    if (threadIdx.x < 16) {
        unsigned char u = upd[base + threadIdx.x];
        upds[threadIdx.x] = u;
        if (u) atomicOr(&anyupd, 1);
    }
    __syncthreads();
    if (!anyupd) return;                   // no score writes in this tile
    for (int idx = threadIdx.x; idx < 16 * 128; idx += 128) {
        int r = idx >> 7, k = idx & 127;
        int node = base + r;
        int bb = node / NNODES;
        float v = (k < 64) ? hidden[node * DD + k] : rel_q[bb * DD + (k - 64)];
        A1[r][k] = (_Float16)v;
    }
    __syncthreads();
    int lane = threadIdx.x & 31, wv = threadIdx.x >> 5;
    int hi = lane >> 4, mrow = lane & 15;
    // phase 1: heuristic = [h|q] @ linW + linb   (K=128, N=64)
    {
        int nn = wv * 16 + (lane & 15);
        v8f c = {};
        for (int kc = 0; kc < 4; ++kc) {
            int K0 = kc * 32;
            v16h a;
#pragma unroll
            for (int v = 0; v < 8; ++v) {
                int ka = K0 + ((v < 4) ? 2 * v : 16 + 2 * (v - 4)) + hi * 8;
                a[2 * v]     = A1[mrow][ka];
                a[2 * v + 1] = A1[mrow][ka + 1];
            }
            v16h w = *(const v16h*)(linWp + (((size_t)wv * 4 + kc) * 32 + lane) * 16);
            c = __builtin_amdgcn_wmma_f32_16x16x32_f16(false, a, false, w,
                                                       (short)0, c, false, false);
        }
        float lb = linb[nn];
#pragma unroll
        for (int r = 0; r < 8; ++r) Heur[r + hi * 8][nn] = c[r] + lb;
    }
    __syncthreads();
    // phase 2: x = hidden * heuristic
    for (int idx = threadIdx.x; idx < 16 * 64; idx += 128) {
        int r = idx >> 6, d = idx & 63;
        int node = base + r;
        A2[r][d] = (_Float16)(hidden[node * DD + d] * Heur[r][d]);
    }
    __syncthreads();
    // phase 3: h1 = relu(x @ W1 + b1)   (K=64, N=128 -> 2 tiles per wave)
    for (int t = 0; t < 2; ++t) {
        int nt = wv * 2 + t;
        int nn2 = nt * 16 + (lane & 15);
        v8f c = {};
        for (int kc = 0; kc < 2; ++kc) {
            int K0 = kc * 32;
            v16h a;
#pragma unroll
            for (int v = 0; v < 8; ++v) {
                int ka = K0 + ((v < 4) ? 2 * v : 16 + 2 * (v - 4)) + hi * 8;
                a[2 * v]     = A2[mrow][ka];
                a[2 * v + 1] = A2[mrow][ka + 1];
            }
            v16h w = *(const v16h*)(W1p + (((size_t)nt * 2 + kc) * 32 + lane) * 16);
            c = __builtin_amdgcn_wmma_f32_16x16x32_f16(false, a, false, w,
                                                       (short)0, c, false, false);
        }
        float bb1 = b1[nn2];
#pragma unroll
        for (int r = 0; r < 8; ++r) H1[r + hi * 8][nn2] = fmaxf(c[r] + bb1, 0.0f);
    }
    __syncthreads();
    // phase 4: out = h1 @ W2 + b2 (dot per row)
    {
        int r = threadIdx.x >> 3, part = threadIdx.x & 7;
        float s = 0.0f;
        for (int k = part * 16; k < part * 16 + 16; ++k) s += H1[r][k] * W2[k];
        red[r][part] = s;
    }
    __syncthreads();
    if (threadIdx.x < 16) {
        int r = threadIdx.x;
        float tot = b2[0];
        for (int p = 0; p < 8; ++p) tot += red[r][p];
        if (upds[r]) score[base + r] = tot;
    }
}

// ---------------- gather output ----------------
__global__ void kgather(const int* __restrict__ t2, const float* __restrict__ score,
                        float* __restrict__ out) {
    int i = threadIdx.x;
    if (i < BQ * 16) out[i] = score[t2[i]];
}

// ---------------- host launcher ----------------
extern "C" void kernel_launch(void* const* d_in, const int* in_sizes, int n_in,
                              void* d_out, int out_size, void* d_ws, size_t ws_size,
                              hipStream_t stream) {
    const int*   h_index       = (const int*)d_in[0];
    const int*   t_index       = (const int*)d_in[1];
    const int*   r_index       = (const int*)d_in[2];
    const float* hidden_states = (const float*)d_in[3];
    // d_in[4] rel_hidden_states: unused by reference
    const float* text_embs     = (const float*)d_in[5];
    const int*   edge_index    = (const int*)d_in[6];
    const int*   edge_attr     = (const int*)d_in[7];
    const int*   all_index     = (const int*)d_in[8];
    const float* rel_embedding = (const float*)d_in[9];
    const float* linear_W      = (const float*)d_in[10];
    const float* linear_b      = (const float*)d_in[11];
    const float* mlp_W1        = (const float*)d_in[12];
    const float* mlp_b1        = (const float*)d_in[13];
    const float* mlp_W2        = (const float*)d_in[14];
    const float* mlp_b2        = (const float*)d_in[15];
    const float* conv_rel_emb  = (const float*)d_in[16];
    const float* conv_W        = (const float*)d_in[17];
    const float* conv_b        = (const float*)d_in[18];
    float* out = (float*)d_out;

    // workspace layout
    char* ws = (char*)d_ws;
    size_t off = 0;
    auto alloc = [&](size_t bytes) -> char* {
        char* p = ws + off;
        off = (off + bytes + 255) & ~(size_t)255;
        return p;
    };
    float*         hidden   = (float*)alloc((size_t)NTOT * DD * 4);
    float*         score    = (float*)alloc((size_t)NTOT * 4);
    float*         li       = (float*)alloc((size_t)NTOT * DD * 4);
    float*         mean_sum = (float*)alloc((size_t)NTOT * DD * 4);
    float*         sq_sum   = (float*)alloc((size_t)NTOT * DD * 4);
    unsigned*      mx_u     = (unsigned*)alloc((size_t)NTOT * DD * 4);
    unsigned*      mn_u     = (unsigned*)alloc((size_t)NTOT * DD * 4);
    float*         cnt_dst  = (float*)alloc((size_t)NTOT * 4);
    float*         cnt_src  = (float*)alloc((size_t)NTOT * 4);
    float*         deg      = (float*)alloc((size_t)NTOT * 4);
    unsigned char* node_sel = (unsigned char*)alloc(NTOT);
    unsigned char* edge_sel = (unsigned char*)alloc(ETOT);
    unsigned char* upd      = (unsigned char*)alloc(NTOT);
    float*         rel_q    = (float*)alloc(BQ * DD * 4);
    int*           h2_0     = (int*)alloc(BQ * 4);
    int*           t2       = (int*)alloc(BQ * 16 * 4);
    float*         s0       = (float*)alloc(BQ * 4);
    float*         sh       = (float*)alloc(BQ * 4);
    unsigned*      nthr     = (unsigned*)alloc(BQ * 4);
    unsigned*      eth      = (unsigned*)alloc(BQ * 4);
    float*         delta    = (float*)alloc(256);
    _Float16*      convWp   = (_Float16*)alloc((size_t)NLAYERS * 13 * DD * DD * 2);
    _Float16*      linWp    = (_Float16*)alloc((size_t)2 * DD * DD * 2);
    _Float16*      W1p      = (_Float16*)alloc((size_t)DD * 2 * DD * 2);
    (void)ws_size; (void)in_sizes; (void)n_in; (void)out_size;

    hipMemsetAsync(hidden, 0, (size_t)NTOT * DD * 4, stream);
    hipMemsetAsync(deg, 0, (size_t)NTOT * 4, stream);

    // one-time weight packs into WMMA-B f16 lane layout
    kpackB<<<(13 * DD * DD + 255) / 256, 256, 0, stream>>>(conv_W, 13 * DD, DD, convWp);
    kpackB<<<(13 * DD * DD + 255) / 256, 256, 0, stream>>>(conv_W + 13 * DD * DD, 13 * DD, DD,
                                                           convWp + 13 * DD * DD);
    kpackB<<<(2 * DD * DD + 255) / 256, 256, 0, stream>>>(linear_W, 2 * DD, DD, linWp);
    kpackB<<<(DD * 2 * DD + 255) / 256, 256, 0, stream>>>(mlp_W1, DD, 2 * DD, W1p);

    kprep<<<BQ, 32, 0, stream>>>(h_index, t_index, r_index, rel_embedding, h2_0, t2, rel_q);
    kscatter_text<<<MTEXT, 64, 0, stream>>>(all_index, text_embs, hidden);
    kscatter_h<<<BQ, 64, 0, stream>>>(h2_0, hidden_states, hidden);
    kinit_sc<<<BQ, 128, 0, stream>>>(hidden_states, rel_q, linear_W, linear_b,
                                     mlp_W1, mlp_b1, mlp_W2, mlp_b2, s0, sh);
    kfill_score<<<(NTOT + 255) / 256, 256, 0, stream>>>(h2_0, s0, sh, score);
    kdeg<<<(ETOT + 255) / 256, 256, 0, stream>>>(edge_index, deg);
    kdelta<<<1, 256, 0, stream>>>(deg, delta);

    for (int l = 0; l < NLAYERS; ++l) {
        knode_topk<<<BQ, 256, 0, stream>>>(score, nthr);
        knode_sel<<<(NTOT + 255) / 256, 256, 0, stream>>>(score, nthr, node_sel);
        kedge_topk<<<BQ, 512, 0, stream>>>(edge_index, node_sel, score, eth);
        kedge_mask<<<(ETOT + 255) / 256, 256, 0, stream>>>(edge_index, node_sel, score,
                                                           eth, edge_sel);
        hipMemsetAsync(mean_sum, 0, (size_t)NTOT * DD * 4, stream);
        hipMemsetAsync(sq_sum, 0, (size_t)NTOT * DD * 4, stream);
        hipMemsetAsync(mx_u, 0x00, (size_t)NTOT * DD * 4, stream);
        hipMemsetAsync(mn_u, 0xFF, (size_t)NTOT * DD * 4, stream);
        hipMemsetAsync(cnt_dst, 0, (size_t)NTOT * 4, stream);
        hipMemsetAsync(cnt_src, 0, (size_t)NTOT * 4, stream);
        klayer_input<<<(NTOT * DD + 255) / 256, 256, 0, stream>>>(score, hidden, li);
        kedge_agg<<<ETOT / 4, 256, 0, stream>>>(edge_index, edge_attr, edge_sel, li,
                                                conv_rel_emb + (size_t)l * 2 * NREL * DD,
                                                mean_sum, sq_sum, mx_u, mn_u,
                                                cnt_dst, cnt_src);
        kupd<<<(NTOT + 255) / 256, 256, 0, stream>>>(cnt_src, cnt_dst, deg, upd);
        kconv<<<NTOT / 16, 128, 0, stream>>>(li, mean_sum, sq_sum, mx_u, mn_u, cnt_dst,
                                             deg, delta, upd,
                                             convWp + (size_t)l * 13 * DD * DD,
                                             conv_b + (size_t)l * DD, hidden);
        kscore<<<NTOT / 16, 128, 0, stream>>>(hidden, rel_q, linWp, linear_b,
                                              W1p, mlp_b1, mlp_W2, mlp_b2, upd, score);
    }
    kgather<<<1, 32, 0, stream>>>(t2, score, out);
}
